// STGCNBlock_15960098472306
// MI455X (gfx1250) — matrix operand order repaired and checked by reading, not compiled, not run on hardware
//
#include <hip/hip_runtime.h>
#include <hip/hip_bf16.h>

#define B_   16
#define T_   12
#define N_   4096
#define CIN  2
#define H_   64
#define E_   65536
#define M_   (B_*T_)           // 192
#define ROWS (M_*N_)           // 786432 rows of 64 channels
#define ROWTILES (ROWS/16)     // 49152

typedef __attribute__((ext_vector_type(16))) __bf16 v16bf;
typedef __attribute__((ext_vector_type(8)))  float  v8f;

union frag16 {
  v16bf v;
  uint4 q[2];
  unsigned short s[16];
};

__device__ __forceinline__ unsigned short f2bf(float f) {
  union { float f; unsigned u; } x; x.f = f;
  unsigned u = x.u + 0x7FFFu + ((x.u >> 16) & 1u);   // round-to-nearest-even
  return (unsigned short)(u >> 16);
}
__device__ __forceinline__ float bf2f(unsigned short u) {
  union { unsigned u; float f; } x; x.u = ((unsigned)u) << 16;  // exact
  return x.f;
}

// ---------------------------------------------------------------------------
// Stage 1: temporal GLU conv, Cin=2 -> H=64 (K=6, VALU; memory bound).
// Output bf16 (feeds WMMA + spmv; halves store traffic).
// ---------------------------------------------------------------------------
__global__ void k_temporal1(const float* __restrict__ x,
                            const float* __restrict__ w1, const float* __restrict__ b1,
                            const float* __restrict__ w2, const float* __restrict__ b2,
                            const float* __restrict__ w3, const float* __restrict__ b3,
                            unsigned short* __restrict__ t1)
{
  __shared__ float sw[3][H_*CIN*3];
  __shared__ float sb[3][H_];
  int tid = threadIdx.x;
  for (int i = tid; i < H_*CIN*3; i += 256) { sw[0][i]=w1[i]; sw[1][i]=w2[i]; sw[2][i]=w3[i]; }
  if (tid < H_) { sb[0][tid]=b1[tid]; sb[1][tid]=b2[tid]; sb[2][tid]=b3[tid]; }
  __syncthreads();

  int c  = tid & 63;
  int nl = tid >> 6;
  long long row = (long long)blockIdx.x*4 + nl;   // row = m*N + n
  int m = (int)(row >> 12);
  int n = (int)(row & 4095);
  int b = m / T_, t = m % T_;

  float xv[3][CIN];
  #pragma unroll
  for (int dt = 0; dt < 3; ++dt) {
    int ts = t + dt - 1;
    bool ok = (ts >= 0 && ts < T_);
    long long base = ((long long)(b*T_ + ts)*N_ + n)*CIN;
    #pragma unroll
    for (int ci = 0; ci < CIN; ++ci) xv[dt][ci] = ok ? x[base + ci] : 0.f;
  }
  float p = sb[0][c], q = sb[1][c], r = sb[2][c];
  #pragma unroll
  for (int ci = 0; ci < CIN; ++ci)
    #pragma unroll
    for (int dt = 0; dt < 3; ++dt) {
      float xx = xv[dt][ci];
      p += sw[0][c*6 + ci*3 + dt]*xx;
      q += sw[1][c*6 + ci*3 + dt]*xx;
      r += sw[2][c*6 + ci*3 + dt]*xx;
    }
  float sg = 1.f/(1.f + __expf(-q));
  float h = p*sg + r;
  t1[row*64 + c] = f2bf(h > 0.f ? h : 0.f);
}

// ---------------------------------------------------------------------------
// Graph normalization + CSR build (counting sort by destination)
// ---------------------------------------------------------------------------
__global__ void k_deg(const int* __restrict__ src, const float* __restrict__ ea,
                      float* __restrict__ deg) {
  int e = blockIdx.x*256 + threadIdx.x;
  atomicAdd(&deg[src[e]], ea[e]);
}
__global__ void k_dis(const float* __restrict__ deg, float* __restrict__ dis) {
  int n = blockIdx.x*256 + threadIdx.x;
  float d = deg[n];
  dis[n] = d > 0.f ? rsqrtf(d) : 0.f;
}
__global__ void k_hist(const int* __restrict__ dst, int* __restrict__ hist) {
  int e = blockIdx.x*256 + threadIdx.x;
  atomicAdd(&hist[dst[e]], 1);
}
__global__ void k_scan(const int* __restrict__ hist, int* __restrict__ rowptr) {
  __shared__ int s[256];
  int tid = threadIdx.x;
  int loc[16]; int sum = 0;
  #pragma unroll
  for (int i = 0; i < 16; ++i) { loc[i] = hist[tid*16 + i]; sum += loc[i]; }
  s[tid] = sum; __syncthreads();
  for (int d = 1; d < 256; d <<= 1) {
    int v = (tid >= d) ? s[tid - d] : 0;
    __syncthreads();
    s[tid] += v;
    __syncthreads();
  }
  int run = (tid == 0) ? 0 : s[tid - 1];
  #pragma unroll
  for (int i = 0; i < 16; ++i) { rowptr[tid*16 + i] = run; run += loc[i]; }
  if (tid == 255) rowptr[4096] = run;
}
__global__ void k_scatter(const int* __restrict__ src, const int* __restrict__ dst,
                          const float* __restrict__ ea, const float* __restrict__ dis,
                          const int* __restrict__ rowptr, int* __restrict__ cursor,
                          int* __restrict__ srcS, float* __restrict__ wnS) {
  int e = blockIdx.x*256 + threadIdx.x;
  int s = src[e], d = dst[e];
  int pos = rowptr[d] + atomicAdd(&cursor[d], 1);
  srcS[pos] = s;
  wnS[pos] = -dis[s]*ea[e]*dis[d];
}

// ---------------------------------------------------------------------------
// spmv over bf16 node features, f32 accumulate, bf16 out:
// out[m,n,c] = alpha * sum_e wn[e]*in[m,src[e],c] + beta*other[m,n,c]
// 256 thr = 4 nodes x 64 channels; edge list is wave-uniform.
// ---------------------------------------------------------------------------
__global__ void k_spmv(const unsigned short* __restrict__ in,
                       const unsigned short* __restrict__ other,
                       const int* __restrict__ rowptr,
                       const int* __restrict__ srcS, const float* __restrict__ wnS,
                       float alpha, float beta,
                       unsigned short* __restrict__ out)
{
  int tid = threadIdx.x;
  int c  = tid & 63;
  int nl = tid >> 6;
  int m = blockIdx.x >> 10;
  int n = ((blockIdx.x & 1023) << 2) + nl;
  const unsigned short* base = in + (long long)m*N_*64;
  int e0 = rowptr[n], e1 = rowptr[n + 1];
  float acc = 0.f;
  for (int e = e0; e < e1; ++e) {
    int s = srcS[e];
    float w = wnS[e];
    acc += w * bf2f(base[(long long)s*64 + c]);
  }
  long long oi = ((long long)m*N_ + n)*64 + c;
  float v = alpha*acc + (beta != 0.f ? beta*bf2f(other[oi]) : 0.f);
  out[oi] = f2bf(v);
}

// ---------------------------------------------------------------------------
// Pre-pack B-matrix WMMA fragments (bf16) in per-lane CDNA5 layout.
// cheb: 4 ntiles x 6 ksteps -> [192 x 64];  t2: 12 ctiles x 6 ksteps -> [192 x 192]
// B element (lane, e): K = kstep*32 + (e&7) + 16*(e>=8) + 8*(lane>=16), col = tile*16 + (lane&15)
// ---------------------------------------------------------------------------
__global__ void k_prep_bfrags(const float* __restrict__ chebW,
                              const float* __restrict__ w1, const float* __restrict__ w2,
                              const float* __restrict__ w3,
                              unsigned short* __restrict__ chebF,
                              unsigned short* __restrict__ t2F)
{
  int idx = blockIdx.x*256 + threadIdx.x;
  if (idx < 12288) {
    int e = idx & 15, lane = (idx >> 4) & 31, ks = (idx >> 9) % 6, nt = idx / (16*32*6);
    int K = ks*32 + (e & 7) + ((e >= 8) ? 16 : 0) + ((lane >= 16) ? 8 : 0);
    int col = nt*16 + (lane & 15);
    int kc = K >> 6, cin = K & 63;
    chebF[idx] = f2bf(chebW[kc*4096 + cin*64 + col]);  // einsum W[k][c][d]
  } else {
    int t = idx - 12288;
    if (t < 36864) {
      int e = t & 15, lane = (t >> 4) & 31, ks = (t >> 9) % 6, j = t / (16*32*6);
      int K = ks*32 + (e & 7) + ((e >= 8) ? 16 : 0) + ((lane >= 16) ? 8 : 0);
      int conv = j >> 2;                        // 0=p,1=q,2=r
      int cout = (j & 3)*16 + (lane & 15);
      int cin = K & 63, dt = K >> 6;
      const float* w = (conv == 0) ? w1 : ((conv == 1) ? w2 : w3);
      t2F[t] = f2bf(w[cout*192 + cin*3 + dt]); // w[o][i][0][dt]
    }
  }
}

// A-fragment loader for bf16 [row][64] feature arrays.
// A layout per lane: e=0..7 -> ch base+0..7, e=8..15 -> ch base+16..23,
// base = (kstep&1)*32 + 8*(lane>=16).
__device__ __forceinline__ frag16 load_afrag(const unsigned short* __restrict__ rowp) {
  frag16 a;
  a.q[0] = *(const uint4*)(rowp);
  a.q[1] = *(const uint4*)(rowp + 16);
  return a;
}

// ---------------------------------------------------------------------------
// Cheb combine: out = relu(Tx0*W0 + Tx1*W1 + Tx2*W2 + b) as one K=192 matmul.
// One wave = one 16-row tile x 64 cols; 24 v_wmma per wave. Pure bf16 loads.
// ---------------------------------------------------------------------------
__global__ void k_cheb_wmma(const unsigned short* __restrict__ Tx0,
                            const unsigned short* __restrict__ Tx1,
                            const unsigned short* __restrict__ Tx2,
                            const unsigned short* __restrict__ chebF,
                            const float* __restrict__ bias,
                            unsigned short* __restrict__ g)
{
  int wave = threadIdx.x >> 5;
  int lane = threadIdx.x & 31;
  int hi = (lane >= 16) ? 1 : 0;
  int lo = lane & 15;
  long long rowBase = (long long)(blockIdx.x*8 + wave)*16;
  const uint4* fb = (const uint4*)chebF;

  v8f zero = {0,0,0,0,0,0,0,0};
  v8f acc[4];
  #pragma unroll
  for (int nt = 0; nt < 4; ++nt) acc[nt] = zero;

  #pragma unroll
  for (int ks = 0; ks < 6; ++ks) {
    const unsigned short* Tx = (ks < 2) ? Tx0 : ((ks < 4) ? Tx1 : Tx2);
    frag16 a = load_afrag(Tx + (rowBase + lo)*64 + (ks & 1)*32 + hi*8);
    #pragma unroll
    for (int nt = 0; nt < 4; ++nt) {
      frag16 bfr;
      int fi = (nt*6 + ks)*32 + lane;
      bfr.q[0] = fb[fi*2]; bfr.q[1] = fb[fi*2 + 1];
      acc[nt] = __builtin_amdgcn_wmma_f32_16x16x32_bf16(
          false, a.v, false, bfr.v, (short)0, acc[nt], false, false);
    }
  }
  #pragma unroll
  for (int nt = 0; nt < 4; ++nt) {
    int col = nt*16 + lo;
    float bv = bias[col];
    #pragma unroll
    for (int r = 0; r < 8; ++r) {
      long long row = rowBase + r + hi*8;   // C layout: M = r + 8*hi, N = lo
      float v = acc[nt][r] + bv;
      v = v > 0.f ? v : 0.f;
      g[row*64 + col] = f2bf(v);
    }
  }
}

// ---------------------------------------------------------------------------
// Temporal GLU conv 2 via WMMA: [16 x 192] x [192 x 192] per wave (p|q|r
// stacked as 12 column tiles), fused sigmoid-gate + relu epilogue. 72 v_wmma.
// ---------------------------------------------------------------------------
__global__ void k_temporal2_wmma(const unsigned short* __restrict__ g,
                                 const unsigned short* __restrict__ t2F,
                                 const float* __restrict__ b1,
                                 const float* __restrict__ b2,
                                 const float* __restrict__ b3,
                                 float* __restrict__ out)
{
  int wave = threadIdx.x >> 5;
  int lane = threadIdx.x & 31;
  int hi = (lane >= 16) ? 1 : 0;
  int lo = lane & 15;
  long long rowBase = (long long)(blockIdx.x*8 + wave)*16;
  int bt = (int)(rowBase >> 12);
  int nb = (int)(rowBase & 4095);
  int b = bt / T_, t = bt % T_;
  const uint4* fb = (const uint4*)t2F;

  v8f zero = {0,0,0,0,0,0,0,0};
  v8f acc[12];
  #pragma unroll
  for (int j = 0; j < 12; ++j) acc[j] = zero;

  #pragma unroll
  for (int ks = 0; ks < 6; ++ks) {
    int dt = ks >> 1;
    int ts = t + dt - 1;
    if (ts < 0 || ts >= T_) continue;       // wave-uniform zero-pad skip
    long long srow = (long long)(b*T_ + ts)*N_ + nb + lo;
    frag16 a = load_afrag(g + srow*64 + (ks & 1)*32 + hi*8);
    #pragma unroll
    for (int j = 0; j < 12; ++j) {
      frag16 bfr;
      int fi = (j*6 + ks)*32 + lane;
      bfr.q[0] = fb[fi*2]; bfr.q[1] = fb[fi*2 + 1];
      acc[j] = __builtin_amdgcn_wmma_f32_16x16x32_bf16(
          false, a.v, false, bfr.v, (short)0, acc[j], false, false);
    }
  }
  #pragma unroll
  for (int jj = 0; jj < 4; ++jj) {
    int cout = jj*16 + lo;
    float bp = b1[cout], bq = b2[cout], br = b3[cout];
    #pragma unroll
    for (int r = 0; r < 8; ++r) {
      long long row = rowBase + r + hi*8;
      float pv = acc[jj][r]     + bp;
      float qv = acc[4 + jj][r] + bq;
      float rv = acc[8 + jj][r] + br;
      float sg = 1.f/(1.f + __expf(-qv));
      float h = pv*sg + rv;
      out[row*64 + cout] = h > 0.f ? h : 0.f;
    }
  }
}

// ---------------------------------------------------------------------------
// BatchNorm (training, batch stats): stats reduce -> finalize -> apply
// ---------------------------------------------------------------------------
__global__ void k_bn_stats(const float* __restrict__ pre, float* __restrict__ sums) {
  __shared__ float sS[64], sQ[64];
  int tid = threadIdx.x;
  if (tid < 64) { sS[tid] = 0.f; sQ[tid] = 0.f; }
  __syncthreads();
  int c = tid & 63;
  int row0 = (blockIdx.x*256 + tid) >> 6;
  const int rstride = (2048*256) >> 6;   // 8192
  float s = 0.f, q = 0.f;
  for (long long row = row0; row < ROWS; row += rstride) {
    float v = pre[row*64 + c];
    s += v; q += v*v;
  }
  atomicAdd(&sS[c], s);
  atomicAdd(&sQ[c], q);
  __syncthreads();
  if (tid < 64) { atomicAdd(&sums[tid], sS[tid]); atomicAdd(&sums[64 + tid], sQ[tid]); }
}
__global__ void k_bn_final(const float* __restrict__ sums, const float* __restrict__ gamma,
                           const float* __restrict__ beta, float* __restrict__ sc) {
  int c = threadIdx.x;
  if (c < 64) {
    const float cnt = (float)ROWS;
    float mu  = sums[c]/cnt;
    float var = sums[64 + c]/cnt - mu*mu;   // biased variance
    float inv = rsqrtf(var + 1e-5f);
    float scale = gamma[c]*inv;
    sc[c]      = scale;
    sc[64 + c] = beta[c] - mu*scale;
  }
}
__global__ void k_bn_apply(const float* __restrict__ pre, const float* __restrict__ sc,
                           float* __restrict__ out) {
  long long i4 = ((long long)blockIdx.x*256 + threadIdx.x)*4;
  float4 v = *(const float4*)(pre + i4);
  int c = (int)(i4 & 63);
  float4 o;
  o.x = v.x*sc[c + 0] + sc[64 + c + 0];
  o.y = v.y*sc[c + 1] + sc[64 + c + 1];
  o.z = v.z*sc[c + 2] + sc[64 + c + 2];
  o.w = v.w*sc[c + 3] + sc[64 + c + 3];
  *(float4*)(out + i4) = o;
}

// ---------------------------------------------------------------------------
extern "C" void kernel_launch(void* const* d_in, const int* in_sizes, int n_in,
                              void* d_out, int out_size, void* d_ws, size_t ws_size,
                              hipStream_t stream)
{
  (void)in_sizes; (void)n_in; (void)out_size; (void)ws_size;
  const float* x     = (const float*)d_in[0];
  const int*   ei    = (const int*)  d_in[1];
  const float* ea    = (const float*)d_in[2];
  const float* t1w1  = (const float*)d_in[3];  const float* t1b1 = (const float*)d_in[4];
  const float* t1w2  = (const float*)d_in[5];  const float* t1b2 = (const float*)d_in[6];
  const float* t1w3  = (const float*)d_in[7];  const float* t1b3 = (const float*)d_in[8];
  const float* chebW = (const float*)d_in[9];  const float* chebB = (const float*)d_in[10];
  const float* t2w1  = (const float*)d_in[11]; const float* t2b1 = (const float*)d_in[12];
  const float* t2w2  = (const float*)d_in[13]; const float* t2b2 = (const float*)d_in[14];
  const float* t2w3  = (const float*)d_in[15]; const float* t2b3 = (const float*)d_in[16];
  const float* bng   = (const float*)d_in[17]; const float* bnb  = (const float*)d_in[18];
  const int* src = ei;
  const int* dst = ei + E_;

  char* ws = (char*)d_ws;
  size_t off = 0;
  auto carve = [&](size_t bytes) -> void* {
    void* p = ws + off;
    off += (bytes + 255) & ~(size_t)255;
    return p;
  };
  const size_t bigH = (size_t)ROWS*64*2;                    // 100 MB bf16
  unsigned short* t1     = (unsigned short*)carve(bigH);    // Tx0 (bf16)
  unsigned short* Tx1    = (unsigned short*)carve(bigH);
  unsigned short* Tx2    = (unsigned short*)carve(bigH);
  unsigned short* g      = (unsigned short*)carve(bigH);
  float*          preOut = (float*)carve((size_t)ROWS*64*4);// 201 MB f32 pre-BN
  float*          deg    = (float*)carve(N_*4);
  float*          dis    = (float*)carve(N_*4);
  int*            hist   = (int*)  carve(N_*4);
  int*            rowptr = (int*)  carve((N_+1)*4);
  int*            cursor = (int*)  carve(N_*4);
  int*            srcS   = (int*)  carve(E_*4);
  float*          wnS    = (float*)carve(E_*4);
  unsigned short* chebF  = (unsigned short*)carve(12288*2);
  unsigned short* t2F    = (unsigned short*)carve(36864*2);
  float*          bnSums = (float*)carve(128*4);
  float*          bnSc   = (float*)carve(128*4);

  hipMemsetAsync(deg,    0, N_*4,  stream);
  hipMemsetAsync(hist,   0, N_*4,  stream);
  hipMemsetAsync(cursor, 0, N_*4,  stream);
  hipMemsetAsync(bnSums, 0, 128*4, stream);

  k_temporal1<<<ROWS/4, 256, 0, stream>>>(x, t1w1, t1b1, t1w2, t1b2, t1w3, t1b3, t1);

  k_deg    <<<E_/256, 256, 0, stream>>>(src, ea, deg);
  k_dis    <<<N_/256, 256, 0, stream>>>(deg, dis);
  k_hist   <<<E_/256, 256, 0, stream>>>(dst, hist);
  k_scan   <<<1,      256, 0, stream>>>(hist, rowptr);
  k_scatter<<<E_/256, 256, 0, stream>>>(src, dst, ea, dis, rowptr, cursor, srcS, wnS);
  k_prep_bfrags<<<192, 256, 0, stream>>>(chebW, t2w1, t2w2, t2w3, chebF, t2F);

  // Chebyshev recurrence: Tx1 = L*Tx0 ; Tx2 = 2*L*Tx1 - Tx0
  k_spmv<<<M_*(N_/4), 256, 0, stream>>>(t1,  t1, rowptr, srcS, wnS, 1.f,  0.f, Tx1);
  k_spmv<<<M_*(N_/4), 256, 0, stream>>>(Tx1, t1, rowptr, srcS, wnS, 2.f, -1.f, Tx2);

  k_cheb_wmma<<<ROWTILES/8, 256, 0, stream>>>(t1, Tx1, Tx2, chebF, chebB, g);

  k_temporal2_wmma<<<ROWTILES/8, 256, 0, stream>>>(g, t2F, t2b1, t2b2, t2b3, preOut);

  k_bn_stats<<<2048, 256, 0, stream>>>(preOut, bnSums);
  k_bn_final<<<1, 64, 0, stream>>>(bnSums, bng, bnb, bnSc);
  k_bn_apply<<<(ROWS*64/4)/256, 256, 0, stream>>>(preOut, bnSc, (float*)d_out);
}
